// TransformerDecoderLayer_11476152615020
// MI455X (gfx1250) — compile-verified, matrix operand order checked by reading, not run on hardware
//
#include <hip/hip_runtime.h>
#include <hip/hip_bf16.h>
#include <stdint.h>

// ---------------------------------------------------------------------------
// CDNA5 (gfx1250) transformer decoder layer.
//  * all GEMM/attention contractions: v_wmma_f32_16x16x32_bf16 (fp32 accum)
//  * bf16 activations between kernels; weights converted+transposed once
//  * GEMM tile staging: TDM tensor_load_to_lds (TENSORcnt), HW row padding
//  * flash attention K staging: global_load_async_to_lds_b128 (ASYNCcnt)
//  * softmax / layernorm statistics in fp32
// ---------------------------------------------------------------------------

typedef __attribute__((ext_vector_type(16))) __bf16 bf16x16;
typedef __attribute__((ext_vector_type(8)))  __bf16 bf16x8;
typedef __attribute__((ext_vector_type(8)))  float  floatx8;
typedef __attribute__((ext_vector_type(4)))  unsigned int u32x4;
typedef __attribute__((ext_vector_type(8)))  int i32x8;
typedef __attribute__((ext_vector_type(4)))  int i32x4;

#define LDT 40   // LDS row stride in bf16 elements (80B rows, 16B-aligned chunks)
#define LKT 72   // LDS stride for 64-wide K rows (144B, 16B-aligned)

static __device__ inline floatx8 wmma_bf16(bf16x16 a, bf16x16 b, floatx8 c) {
  return __builtin_amdgcn_wmma_f32_16x16x32_bf16(false, a, false, b, (short)0, c,
                                                 false, false);
}

// 16x32 A/B fragment from a bf16 row (LDS or global).
// Lane l (hf = l>>4): elems 0-7 = k[hf*8 .. +7], 8-15 = k[16+hf*8 .. +7].
static __device__ inline bf16x16 load_frag_b(const __bf16* rowbase, int hf) {
  bf16x8 lo = *(const bf16x8*)(rowbase + hf * 8);
  bf16x8 hi = *(const bf16x8*)(rowbase + 16 + hf * 8);
  return __builtin_shufflevector(lo, hi, 0, 1, 2, 3, 4, 5, 6, 7,
                                 8, 9, 10, 11, 12, 13, 14, 15);
}

// Async 16-byte global -> LDS copy (gfx1250 async path, ASYNCcnt).
static __device__ inline void async_copy16(void* lds_dst, const void* gsrc) {
  asm volatile("global_load_async_to_lds_b128 %0, %1, off"
               :
               : "v"((unsigned)(uintptr_t)lds_dst),
                 "v"((unsigned long long)(uintptr_t)gsrc)
               : "memory");
}
static __device__ inline void async_wait0() {
  asm volatile("s_wait_asynccnt 0" ::: "memory");
}

// ---------------------------------------------------------------------------
// TDM: DMA a 128-row x 32-element bf16 tile (row stride lda elements) from
// global memory into LDS with hardware padding: 64B of data then 16B pad per
// row -> 80B LDS row stride (= LDT elements).  ISA 8.3/8.4 descriptor layout.
// ---------------------------------------------------------------------------
static __device__ inline void tdm_load_tile_128x32(const __bf16* gtile,
                                                   unsigned lda_elems,
                                                   void* lds_dst) {
  unsigned long long ga = (unsigned long long)(uintptr_t)gtile;
  u32x4 g0;
  g0[0] = 1u;                                    // count=1, user descriptor
  g0[1] = (unsigned)(uintptr_t)lds_dst;          // lds_addr
  g0[2] = (unsigned)(ga & 0xFFFFFFFFu);          // global_addr[31:0]
  g0[3] = (unsigned)((ga >> 32) & 0x01FFFFFFu)   // global_addr[56:32]
          | (2u << 30);                          // type = 2 ("image")
  i32x8 g1;
  g1[0] = (1 << 16)        // data_size = 2 bytes
        | (1 << 20)        // pad_enable
        | (3 << 22)        // pad_interval: 16 DWORDs (64B) between pads
        | (3 << 25);       // pad_amount:   4 DWORDs (16B)
  g1[1] = (int)((lda_elems & 0xFFFFu) << 16);          // tensor_dim0[15:0]
  g1[2] = (int)((lda_elems >> 16) & 0xFFFFu);          // tensor_dim0[31:16] ;
                                                       // tensor_dim1[15:0]=0
  g1[3] = 0x10 | (32 << 16);   // tensor_dim1[31:16]=0x10 (1M rows), tile_dim0=32
  g1[4] = 128;                 // tile_dim1 = 128 rows, tile_dim2 = 0
  g1[5] = (int)lda_elems;      // tensor_dim0_stride[31:0]
  g1[6] = 0;                   // stride0[47:32]=0, stride1[15:0]=0
  g1[7] = 0;                   // stride1[47:16]=0
  i32x4 gz4 = {0, 0, 0, 0};          // groups 2/3 unused (2-D tensor)
  i32x8 gz8 = {0, 0, 0, 0, 0, 0, 0, 0};
  __builtin_amdgcn_tensor_load_to_lds(g0, g1, gz4, gz4, gz8, 0);
}
static __device__ inline void tensor_wait0() {
  __builtin_amdgcn_s_wait_tensorcnt(0);
}

// ---------------------------------------------------------------------------
// fp32 -> bf16 elementwise convert (4 elements / thread).
// ---------------------------------------------------------------------------
__global__ __launch_bounds__(256) void cvt_bf16(const float* __restrict__ in,
                                                __bf16* __restrict__ out,
                                                unsigned n) {
  unsigned i = (blockIdx.x * 256u + threadIdx.x) * 4u;
  if (i + 3 < n) {
    float4 f = *(const float4*)(in + i);
    out[i + 0] = (__bf16)f.x;
    out[i + 1] = (__bf16)f.y;
    out[i + 2] = (__bf16)f.z;
    out[i + 3] = (__bf16)f.w;
  }
}

// ---------------------------------------------------------------------------
// fp32 [K][N] -> bf16 [N][K] convert + transpose, 32x32 tiles via LDS.
// ---------------------------------------------------------------------------
__global__ __launch_bounds__(256) void cvt_transpose_bf16(
    const float* __restrict__ in, __bf16* __restrict__ outT, int K, int N) {
  __shared__ float t[32][33];
  int n0 = blockIdx.x * 32, k0 = blockIdx.y * 32;
  int tx = threadIdx.x & 31, ty = threadIdx.x >> 5;  // 32 x 8
#pragma unroll
  for (int i = 0; i < 4; ++i)
    t[ty * 4 + i][tx] = in[(unsigned)(k0 + ty * 4 + i) * N + n0 + tx];
  __syncthreads();
#pragma unroll
  for (int i = 0; i < 4; ++i)
    outT[(unsigned)(n0 + ty * 4 + i) * K + k0 + tx] = (__bf16)t[tx][ty * 4 + i];
}

// ---------------------------------------------------------------------------
// Fused residual-add + LayerNorm (+ optional row multiplier).  D = 512.
// Writes fp32 residual accumulator (optional) and bf16 normalized output.
// ---------------------------------------------------------------------------
__global__ __launch_bounds__(256) void ln_fused(
    const float* __restrict__ a, const float* __restrict__ b_add,
    const float* __restrict__ g, const float* __restrict__ beta,
    const float* __restrict__ rowmul, float* __restrict__ sum_out,
    __bf16* __restrict__ out) {
  const int D = 512;
  unsigned row = blockIdx.x;
  unsigned tid = threadIdx.x;
  unsigned base = row * D;

  float x0 = a[base + tid];
  float x1 = a[base + tid + 256];
  if (b_add) {
    x0 += b_add[base + tid];
    x1 += b_add[base + tid + 256];
  }
  if (sum_out) {
    sum_out[base + tid] = x0;
    sum_out[base + tid + 256] = x1;
  }

  __shared__ float red1[8];
  __shared__ float red2[8];
  int wave = tid >> 5, lane = tid & 31;

  float s = x0 + x1;
#pragma unroll
  for (int off = 16; off; off >>= 1) s += __shfl_xor(s, off, 32);
  if (lane == 0) red1[wave] = s;
  __syncthreads();
  float tot = 0.0f;
#pragma unroll
  for (int i = 0; i < 8; ++i) tot += red1[i];
  float mean = tot * (1.0f / D);

  float d0 = x0 - mean, d1 = x1 - mean;
  float s2 = d0 * d0 + d1 * d1;
#pragma unroll
  for (int off = 16; off; off >>= 1) s2 += __shfl_xor(s2, off, 32);
  if (lane == 0) red2[wave] = s2;
  __syncthreads();
  float tot2 = 0.0f;
#pragma unroll
  for (int i = 0; i < 8; ++i) tot2 += red2[i];
  float rstd = rsqrtf(tot2 * (1.0f / D) + 1e-5f);

  float mul = rowmul ? rowmul[row] : 1.0f;
  out[base + tid]       = (__bf16)((d0 * rstd * g[tid] + beta[tid]) * mul);
  out[base + tid + 256] = (__bf16)((d1 * rstd * g[tid + 256] + beta[tid + 256]) * mul);
}

// ---------------------------------------------------------------------------
// Tiled WMMA GEMM:  C = epilogue( A(MxK bf16, row) * Bt(NxK bf16, row)^T )
// Block tile 128x128, K-step 32, 8 waves (4x2), wave tile 32x64 (2x4 WMMAs).
// Tiles staged by the Tensor Data Mover (wave 0 issues 2 descriptors/step).
// Epilogue compile-time specialized.
// ---------------------------------------------------------------------------
template <int RELU, int RES, int RS, int OBF16>
__global__ __launch_bounds__(256) void gemm_wmma(
    const __bf16* __restrict__ A, int lda, const __bf16* __restrict__ Bt,
    int ldb, const float* __restrict__ bias, const float* __restrict__ resid,
    const float* __restrict__ rowscale, float* __restrict__ Cf,
    __bf16* __restrict__ Cb, int ldc, int K, float alpha) {
  __shared__ __bf16 lA[128 * LDT];
  __shared__ __bf16 lB[128 * LDT];

  unsigned tid = threadIdx.x;
  unsigned m0 = blockIdx.y * 128, n0 = blockIdx.x * 128;
  int wave = tid >> 5, lane = tid & 31;
  int wm = wave & 3, wn = wave >> 2;  // 4 M-slices x 2 N-slices
  int hf = lane >> 4, l16 = lane & 15;

  floatx8 zero = {0, 0, 0, 0, 0, 0, 0, 0};
  floatx8 acc[2][4];
#pragma unroll
  for (int mt = 0; mt < 2; ++mt)
#pragma unroll
    for (int nt = 0; nt < 4; ++nt) acc[mt][nt] = zero;

  for (unsigned kk = 0; kk < (unsigned)K; kk += 32) {
    // ---- TDM staging of both 128x32 tiles (wave 0 only, EXEC-independent) --
    if (wave == 0) {
      tdm_load_tile_128x32(A + m0 * (unsigned)lda + kk, (unsigned)lda, &lA[0]);
      tdm_load_tile_128x32(Bt + n0 * (unsigned)ldb + kk, (unsigned)ldb, &lB[0]);
      tensor_wait0();
    }
    __syncthreads();

    bf16x16 af[2], bfr[4];
#pragma unroll
    for (int mt = 0; mt < 2; ++mt)
      af[mt] = load_frag_b(&lA[(wm * 32 + mt * 16 + l16) * LDT], hf);
#pragma unroll
    for (int nt = 0; nt < 4; ++nt)
      bfr[nt] = load_frag_b(&lB[(wn * 64 + nt * 16 + l16) * LDT], hf);
#pragma unroll
    for (int mt = 0; mt < 2; ++mt)
#pragma unroll
      for (int nt = 0; nt < 4; ++nt)
        acc[mt][nt] = wmma_bf16(af[mt], bfr[nt], acc[mt][nt]);
    __syncthreads();
  }

  // Epilogue.  C/D layout: lane l: n = l&15, rows (l>>4)*8 + v.
#pragma unroll
  for (int mt = 0; mt < 2; ++mt) {
#pragma unroll
    for (int nt = 0; nt < 4; ++nt) {
      unsigned n = n0 + wn * 64 + nt * 16 + l16;
      float bv = bias[n];
#pragma unroll
      for (int v = 0; v < 8; ++v) {
        unsigned m = m0 + wm * 32 + mt * 16 + hf * 8 + v;
        float val = (acc[mt][nt][v] + bv) * alpha;
        if (RELU) val = fmaxf(val, 0.0f);
        if (RES) val += resid[m * (unsigned)ldc + n];
        if (RS) val *= rowscale[m];
        if (OBF16)
          Cb[m * (unsigned)ldc + n] = (__bf16)val;
        else
          Cf[m * (unsigned)ldc + n] = val;
      }
    }
  }
}

// ---------------------------------------------------------------------------
// Flash attention.  One wave per (b, h, 16-query tile); the 8 waves of a block
// cover 8 consecutive q-tiles of the SAME (b,h), so K and V tiles are staged
// cooperatively per block:
//   K (32x64, row-major)   via global_load_async_to_lds_b128
//   V^T (64 dh x 32 keys)  via 16B global load + 8 ds_store_b16 per thread
// All WMMA B-fragments then come from contiguous 16B LDS loads.
// ---------------------------------------------------------------------------
#define FAD 512
#define FADH 64

__global__ __launch_bounds__(256) void flash_attn(
    const __bf16* __restrict__ q, const __bf16* __restrict__ kmat,
    const __bf16* __restrict__ vmat, __bf16* __restrict__ out,
    const int* __restrict__ kpm, int QL, int KL, int H, int causal) {
  unsigned tid = threadIdx.x;
  int lane = tid & 31;
  int wslot = tid >> 5;
  int nqt = QL >> 4;
  int gw = blockIdx.x * 8 + wslot;
  int qt = gw % nqt;
  int h = (gw / nqt) % H;
  int b = gw / (nqt * H);
  int qt0 = (blockIdx.x * 8) % nqt;  // block-uniform base q-tile
  int hf = lane >> 4, l16 = lane & 15;

  __shared__ __bf16 lK[32 * LKT];      // K tile  [key][dh]
  __shared__ __bf16 lVT[64 * LDT];     // V^T tile [dh][key]
  __shared__ __bf16 lp[8][16 * LDT];   // per-wave P staging

  const __bf16* qbase = q + (unsigned)(b * QL + qt * 16 + l16) * FAD + h * FADH;
  bf16x16 qf[2];
  qf[0] = load_frag_b(qbase, hf);
  qf[1] = load_frag_b(qbase + 32, hf);

  float mst[8], lst[8];
  floatx8 zero = {0, 0, 0, 0, 0, 0, 0, 0};
  floatx8 oacc[4];
#pragma unroll
  for (int v = 0; v < 8; ++v) {
    mst[v] = -1e30f;
    lst[v] = 0.0f;
  }
#pragma unroll
  for (int t = 0; t < 4; ++t) oacc[t] = zero;

  int kendW = causal ? (qt * 16 + 16) : KL;        // this wave's bound
  int kendB = causal ? (qt0 * 16 + 128) : KL;      // block max bound
  unsigned sr = tid >> 3, sc = tid & 7;            // stage: 32 rows x 8 chunks

  for (int kb = 0; kb < kendB; kb += 32) {
    // ---- cooperative K/V staging ---------------------------------------
    {
      unsigned grow = (unsigned)(b * KL + kb + sr) * FAD + h * FADH + sc * 8;
      async_copy16(&lK[sr * LKT + sc * 8], kmat + grow);
      bf16x8 vv = *(const bf16x8*)(vmat + grow);
#pragma unroll
      for (int i = 0; i < 8; ++i) lVT[(sc * 8 + i) * LDT + sr] = vv[i];
    }
    async_wait0();
    __syncthreads();

    if (kb < kendW) {  // wave-uniform: EXEC stays all-ones inside
      // ---- scores for keys [kb, kb+32): two 16x16 tiles ----------------
      floatx8 s0 = zero, s1 = zero;
      {
        const __bf16* k0 = &lK[l16 * LKT];
        s0 = wmma_bf16(qf[0], load_frag_b(k0, hf), s0);
        s0 = wmma_bf16(qf[1], load_frag_b(k0 + 32, hf), s0);
        const __bf16* k1 = &lK[(16 + l16) * LKT];
        s1 = wmma_bf16(qf[0], load_frag_b(k1, hf), s1);
        s1 = wmma_bf16(qf[1], load_frag_b(k1 + 32, hf), s1);
      }

      int key0 = kb + l16, key1 = kb + 16 + l16;
      bool pk0 = true, pk1 = true;
      if (!causal) {
        pk0 = kpm[b * KL + key0] != 0;
        pk1 = kpm[b * KL + key1] != 0;
      }

      // ---- online softmax (fp32) ---------------------------------------
#pragma unroll
      for (int v = 0; v < 8; ++v) {
        int row = qt * 16 + hf * 8 + v;
        bool ok0 = causal ? (key0 <= row) : pk0;
        bool ok1 = causal ? (key1 <= row) : pk1;
        float sv0 = ok0 ? s0[v] : -1e30f;
        float sv1 = ok1 ? s1[v] : -1e30f;

        float mx = fmaxf(sv0, sv1);
#pragma unroll
        for (int off = 8; off; off >>= 1) mx = fmaxf(mx, __shfl_xor(mx, off, 32));
        float mnew = fmaxf(mst[v], mx);
        float corr = __expf(mst[v] - mnew);
        mst[v] = mnew;

        float p0 = __expf(sv0 - mnew);
        float p1 = __expf(sv1 - mnew);
        float rs = p0 + p1;
#pragma unroll
        for (int off = 8; off; off >>= 1) rs += __shfl_xor(rs, off, 32);
        lst[v] = lst[v] * corr + rs;
#pragma unroll
        for (int t = 0; t < 4; ++t) oacc[t][v] *= corr;

        lp[wslot][(hf * 8 + v) * LDT + l16] = (__bf16)p0;
        lp[wslot][(hf * 8 + v) * LDT + 16 + l16] = (__bf16)p1;
      }
      // Cross-lane LDS transpose within the wave: drain DS pipe explicitly.
      asm volatile("s_wait_dscnt 0" ::: "memory");

      bf16x16 pf = load_frag_b(&lp[wslot][l16 * LDT], hf);

      // ---- P * V (B-fragments from V^T rows in LDS) --------------------
#pragma unroll
      for (int t = 0; t < 4; ++t) {
        bf16x16 vf = load_frag_b(&lVT[(t * 16 + l16) * LDT], hf);
        oacc[t] = wmma_bf16(pf, vf, oacc[t]);
      }
    }
    __syncthreads();
  }

  // ---- normalize and write bf16 ctx ---------------------------------------
#pragma unroll
  for (int t = 0; t < 4; ++t) {
#pragma unroll
    for (int v = 0; v < 8; ++v) {
      int row = qt * 16 + hf * 8 + v;
      out[(unsigned)(b * QL + row) * FAD + h * FADH + t * 16 + l16] =
          (__bf16)(oacc[t][v] / lst[v]);
    }
  }
}

// ---------------------------------------------------------------------------
// Host-side pipeline.
// ---------------------------------------------------------------------------
extern "C" void kernel_launch(void* const* d_in, const int* in_sizes, int n_in,
                              void* d_out, int out_size, void* d_ws,
                              size_t ws_size, hipStream_t stream) {
  (void)in_sizes; (void)n_in; (void)out_size; (void)ws_size;

  const float* tgt  = (const float*)d_in[1];
  const float* mem  = (const float*)d_in[2];
  const float* pos  = (const float*)d_in[3];
  const float* tkpm = (const float*)d_in[5];
  const int*   mkpm = (const int*)d_in[6];
  const float* sa_w[4] = {(const float*)d_in[7], (const float*)d_in[9],
                          (const float*)d_in[11], (const float*)d_in[13]};
  const float* sa_b[4] = {(const float*)d_in[8], (const float*)d_in[10],
                          (const float*)d_in[12], (const float*)d_in[14]};
  const float* ca_w[4] = {(const float*)d_in[15], (const float*)d_in[17],
                          (const float*)d_in[19], (const float*)d_in[21]};
  const float* ca_b[4] = {(const float*)d_in[16], (const float*)d_in[18],
                          (const float*)d_in[20], (const float*)d_in[22]};
  const float* l1w = (const float*)d_in[23]; const float* l1b = (const float*)d_in[24];
  const float* l2w = (const float*)d_in[25]; const float* l2b = (const float*)d_in[26];
  const float* n1g = (const float*)d_in[27]; const float* n1b = (const float*)d_in[28];
  const float* n2g = (const float*)d_in[29]; const float* n2b = (const float*)d_in[30];
  const float* n3g = (const float*)d_in[31]; const float* n3b = (const float*)d_in[32];
  float* out = (float*)d_out;

  const int Bn = 2, QLn = 2048, KLn = 2048, Dn = 512, Hn = 8, FFn = 2048;
  const size_t SZ = (size_t)Bn * QLn * Dn;  // 2,097,152 elements

  // ---- workspace layout (bytes) ----
  char* wsb = (char*)d_ws;
  float*  w_tgt = (float*)wsb;                           // SZ f32   (8 MB)
  __bf16* w_xln = (__bf16*)(wsb + SZ * 4);               // SZ bf16  (4 MB)
  __bf16* w_q   = (__bf16*)(wsb + SZ * 4 + SZ * 2);
  __bf16* w_k   = (__bf16*)(wsb + SZ * 4 + SZ * 4);
  __bf16* w_v   = (__bf16*)(wsb + SZ * 4 + SZ * 6);
  __bf16* w_ctx = (__bf16*)(wsb + SZ * 4 + SZ * 8);
  __bf16* w_ffn = (__bf16*)(wsb + SZ * 4 + SZ * 10);     // 4*SZ bf16 (16 MB)
  __bf16* w_mem = (__bf16*)(wsb + SZ * 4 + SZ * 18);     // SZ bf16
  __bf16* w_wt  = (__bf16*)(wsb + SZ * 4 + SZ * 20);     // transposed weights
  const size_t WSQ = (size_t)Dn * Dn;      // 262144
  __bf16* t_saw[4], *t_caw[4];
  for (int i = 0; i < 4; ++i) t_saw[i] = w_wt + i * WSQ;
  for (int i = 0; i < 4; ++i) t_caw[i] = w_wt + (4 + i) * WSQ;
  __bf16* t_l1w = w_wt + 8 * WSQ;                        // [FF][D]
  __bf16* t_l2w = w_wt + 8 * WSQ + (size_t)Dn * FFn;     // [D][FF]

  int rows = Bn * QLn;  // 4096
  dim3 blk(256);
  dim3 gP(Dn / 128, rows / 128);     // 4 x 32
  dim3 gF1(FFn / 128, rows / 128);   // 16 x 32
  dim3 gT(Dn / 32, Dn / 32);
  dim3 gT1(FFn / 32, Dn / 32);       // lin1: K=D rows, N=FF cols
  dim3 gT2(Dn / 32, FFn / 32);       // lin2: K=FF rows, N=D cols
  int fblocks = Bn * Hn * (QLn / 16) / 8;  // 256 blocks (8 waves each)
  float qscale = 0.125f;  // 1/sqrt(DH=64)

  // ---- one-time (per launch) weight/memory conversion ----
  for (int i = 0; i < 4; ++i) {
    cvt_transpose_bf16<<<gT, blk, 0, stream>>>(sa_w[i], t_saw[i], Dn, Dn);
    cvt_transpose_bf16<<<gT, blk, 0, stream>>>(ca_w[i], t_caw[i], Dn, Dn);
  }
  cvt_transpose_bf16<<<gT1, blk, 0, stream>>>(l1w, t_l1w, Dn, FFn);
  cvt_transpose_bf16<<<gT2, blk, 0, stream>>>(l2w, t_l2w, FFn, Dn);
  cvt_bf16<<<(unsigned)(SZ / 1024), blk, 0, stream>>>(mem, w_mem, (unsigned)SZ);

  // ---- self-attention block ----
  ln_fused<<<rows, blk, 0, stream>>>(tgt, pos, n1g, n1b, tkpm, w_tgt, w_xln);
  gemm_wmma<0, 0, 0, 1><<<gP, blk, 0, stream>>>(w_xln, Dn, t_saw[0], Dn, sa_b[0],
      nullptr, nullptr, nullptr, w_q, Dn, Dn, qscale);
  gemm_wmma<0, 0, 0, 1><<<gP, blk, 0, stream>>>(w_xln, Dn, t_saw[1], Dn, sa_b[1],
      nullptr, nullptr, nullptr, w_k, Dn, Dn, 1.0f);
  gemm_wmma<0, 0, 0, 1><<<gP, blk, 0, stream>>>(w_xln, Dn, t_saw[2], Dn, sa_b[2],
      nullptr, nullptr, nullptr, w_v, Dn, Dn, 1.0f);
  flash_attn<<<fblocks, blk, 0, stream>>>(w_q, w_k, w_v, w_ctx, nullptr, QLn,
                                          KLn, Hn, 1);
  gemm_wmma<0, 1, 0, 0><<<gP, blk, 0, stream>>>(w_ctx, Dn, t_saw[3], Dn, sa_b[3],
      w_tgt, nullptr, w_tgt, nullptr, Dn, Dn, 1.0f);

  // ---- cross-attention block ----
  ln_fused<<<rows, blk, 0, stream>>>(w_tgt, nullptr, n2g, n2b, nullptr, nullptr,
                                     w_xln);
  gemm_wmma<0, 0, 0, 1><<<gP, blk, 0, stream>>>(w_xln, Dn, t_caw[0], Dn, ca_b[0],
      nullptr, nullptr, nullptr, w_q, Dn, Dn, qscale);
  gemm_wmma<0, 0, 0, 1><<<gP, blk, 0, stream>>>(w_mem, Dn, t_caw[1], Dn, ca_b[1],
      nullptr, nullptr, nullptr, w_k, Dn, Dn, 1.0f);
  gemm_wmma<0, 0, 0, 1><<<gP, blk, 0, stream>>>(w_mem, Dn, t_caw[2], Dn, ca_b[2],
      nullptr, nullptr, nullptr, w_v, Dn, Dn, 1.0f);
  flash_attn<<<fblocks, blk, 0, stream>>>(w_q, w_k, w_v, w_ctx, mkpm, QLn, KLn,
                                          Hn, 0);
  gemm_wmma<0, 1, 0, 0><<<gP, blk, 0, stream>>>(w_ctx, Dn, t_caw[3], Dn, ca_b[3],
      w_tgt, nullptr, w_tgt, nullptr, Dn, Dn, 1.0f);

  // ---- feed-forward block ----
  ln_fused<<<rows, blk, 0, stream>>>(w_tgt, nullptr, n3g, n3b, nullptr, nullptr,
                                     w_xln);
  gemm_wmma<1, 0, 0, 1><<<gF1, blk, 0, stream>>>(w_xln, Dn, t_l1w, Dn, l1b,
      nullptr, nullptr, nullptr, w_ffn, FFn, Dn, 1.0f);
  gemm_wmma<0, 1, 1, 0><<<gP, blk, 0, stream>>>(w_ffn, FFn, t_l2w, FFn, l2b,
      w_tgt, tkpm, out, nullptr, Dn, FFn, 1.0f);
}
